// ModelEUI64Gen_37812892074570
// MI455X (gfx1250) — matrix operand order repaired and checked by reading, not compiled, not run on hardware
//
#include <hip/hip_runtime.h>
#include <math.h>

// ---------------------------------------------------------------------------
// MI455X / gfx1250, wave32. WMMA f32 = A(16x32 f16) * B(32x16 f16) + C.
// ---------------------------------------------------------------------------
typedef __attribute__((ext_vector_type(16))) _Float16 v16h;
typedef __attribute__((ext_vector_type(8)))  _Float16 v8h;
typedef __attribute__((ext_vector_type(4)))  _Float16 h4;
typedef __attribute__((ext_vector_type(8)))  float    v8f;
typedef __attribute__((ext_vector_type(4)))  float    v4f;

static __device__ __forceinline__ v8f wmma_f16(v16h a, v16h b, v8f c) {
  return __builtin_amdgcn_wmma_f32_16x16x32_f16(false, a, false, b, (short)0, c,
                                                false, false);
}

static __device__ __forceinline__ h4 cvt4(v4f f) {
  h4 h;
  h[0] = (_Float16)f[0]; h[1] = (_Float16)f[1];
  h[2] = (_Float16)f[2]; h[3] = (_Float16)f[3];
  return h;
}

static __device__ __forceinline__ v8h cvt8(v8f f) {
  v4f lo = __builtin_shufflevector(f, f, 0, 1, 2, 3);
  v4f hi = __builtin_shufflevector(f, f, 4, 5, 6, 7);
  return __builtin_shufflevector(cvt4(lo), cvt4(hi), 0, 1, 2, 3, 4, 5, 6, 7);
}

// 16 halves from two 16B-aligned 8-half chunks.
static __device__ __forceinline__ v16h ld16(const _Float16* lo,
                                            const _Float16* hi) {
  v8h a = *(const v8h*)lo;
  v8h b = *(const v8h*)hi;
  return __builtin_shufflevector(a, b, 0, 1, 2, 3, 4, 5, 6, 7,
                                 8, 9, 10, 11, 12, 13, 14, 15);
}

static __device__ __forceinline__ float gelu_tanh(float x) {
  const float k0 = 0.7978845608028654f;
  float x3 = x * x * x;
  return 0.5f * x * (1.0f + tanhf(k0 * (x + 0.044715f * x3)));
}

// ---------------------------------------------------------------------------
// GEMM: C[M,N] = act(A[M,K] @ B[K,N] + bias) (*rowScale) (+=/=)
// 128x128 macro-tile, BK=32, 256 threads = 8 waves, each 64Mx32N (4x2 WMMA).
// ---------------------------------------------------------------------------
#define BM 128
#define BN 128
#define BK 32

template <bool GN, int ACT, bool SC, bool AC>
__global__ __launch_bounds__(256) void k_gemm(
    const float* __restrict__ A, const float* __restrict__ Bw,
    const float* __restrict__ bias, float* __restrict__ C,
    int M, int N, int K,
    const float* __restrict__ rowScale, int rsStride)
{
  __shared__ __align__(16) _Float16 As[BM][40];   // 80B rows
  __shared__ __align__(16) _Float16 Bs[BK][136];  // 272B rows

  const int tid  = threadIdx.x;
  const int lane = tid & 31;
  const int wave = tid >> 5;
  const int half = lane >> 4;
  const int l16  = lane & 15;
  const int m0 = blockIdx.y * BM;
  const int n0 = blockIdx.x * BN;
  const int wm = (wave >> 2) * 64;   // 2 wave-rows of 64
  const int wn = (wave &  3) * 32;   // 4 wave-cols of 32

  v8f acc[4][2];
  #pragma unroll
  for (int i = 0; i < 4; ++i)
    #pragma unroll
    for (int j = 0; j < 2; ++j)
      #pragma unroll
      for (int r = 0; r < 8; ++r) acc[i][j][r] = 0.0f;

  for (int k0 = 0; k0 < K; k0 += BK) {
    // ---- stage A tile: 128x32 = 1024 float4, 4 per thread
    #pragma unroll
    for (int it = 0; it < 4; ++it) {
      int i  = tid + it * 256;
      int r  = i >> 3, c4 = i & 7;
      v4f v = *(const v4f*)(A + (size_t)(m0 + r) * K + k0 + c4 * 4);
      *(h4*)&As[r][c4 * 4] = cvt4(v);
    }
    // ---- stage B tile: 32x128 = 1024 float4
    if (GN) {
      for (int i = tid; i < BK * BN; i += 256) {
        int r = i >> 7, c = i & 127;
        int gn = n0 + c;
        float v = (gn < N) ? Bw[(size_t)(k0 + r) * N + gn] : 0.0f;
        Bs[r][c] = (_Float16)v;
      }
    } else {
      #pragma unroll
      for (int it = 0; it < 4; ++it) {
        int i  = tid + it * 256;
        int r  = i >> 5, c4 = i & 31;
        v4f v = *(const v4f*)(Bw + (size_t)(k0 + r) * N + n0 + c4 * 4);
        *(h4*)&Bs[r][c4 * 4] = cvt4(v);
      }
    }
    if (k0 + BK < K)
      __builtin_prefetch(&A[(size_t)(m0 + (tid >> 1)) * K + k0 + BK], 0, 1);
    __syncthreads();

    // ---- fragments
    v16h af[4], bf[2];
    #pragma unroll
    for (int i = 0; i < 4; ++i) {
      int row = wm + i * 16 + l16;
      af[i] = ld16(&As[row][half * 8], &As[row][half * 8 + 16]);
    }
    #pragma unroll
    for (int j2 = 0; j2 < 2; ++j2) {
      int col = wn + j2 * 16 + l16;
      int kb  = half * 8;
      #pragma unroll
      for (int j = 0; j < 16; ++j) bf[j2][j] = Bs[kb + (j & 7) + (j >> 3) * 16][col];
    }
    #pragma unroll
    for (int i = 0; i < 4; ++i)
      #pragma unroll
      for (int j = 0; j < 2; ++j)
        acc[i][j] = wmma_f16(af[i], bf[j], acc[i][j]);
    __syncthreads();
  }

  // ---- epilogue
  #pragma unroll
  for (int i = 0; i < 4; ++i) {
    #pragma unroll
    for (int j = 0; j < 2; ++j) {
      int colg = n0 + wn + j * 16 + l16;
      if (GN && colg >= N) continue;
      float bv = bias[colg];
      #pragma unroll
      for (int r = 0; r < 8; ++r) {
        int rowg = m0 + wm + i * 16 + half * 8 + r;
        float v = acc[i][j][r] + bv;
        if (ACT == 1) v = gelu_tanh(v);
        if (SC) v *= rowScale[(size_t)rowg * rsStride];
        size_t idx = (size_t)rowg * N + colg;
        if (AC) C[idx] += v; else C[idx] = v;
      }
    }
  }
}

// ---------------------------------------------------------------------------
// Flash attention, transposed pipeline:
//   S^T[32k,16q] = K @ Q^T   (A = K tile from LDS, B = Q^T in registers)
//   O^T[64d,16q] = V^T @ P^T (A = V^T gather,  B = P^T via LDS restage)
// ---------------------------------------------------------------------------
__global__ __launch_bounds__(128) void k_attn(
    const float* __restrict__ qkv, float* __restrict__ out,
    int S, int Dm, int Hn)
{
  const int DHc  = Dm / Hn;                 // 64
  const int b    = blockIdx.x / Hn;
  const int h    = blockIdx.x % Hn;
  const int qblk = blockIdx.y;
  const int tid  = threadIdx.x;
  const int lane = tid & 31, wave = tid >> 5;
  const int half = lane >> 4, l16 = lane & 15;
  const int q0   = qblk * 64 + wave * 16;
  const int qg   = q0 + l16;
  const size_t row3D = (size_t)3 * Dm;
  const float* qbase = qkv + (size_t)b * S * row3D + h * DHc;
  const float* kbase = qbase + Dm;
  const float* vbase = qbase + 2 * Dm;

  __shared__ __align__(16) _Float16 Ks[32][72];
  __shared__ __align__(16) _Float16 Vs[32][72];
  __shared__ __align__(16) _Float16 PsT[4][16][40];

  v16h bq[2];
  {
    const float* qp = qbase + (size_t)qg * row3D;
    #pragma unroll
    for (int c = 0; c < 2; ++c) {
      int d0 = c * 32 + half * 16;
      h4 a0 = cvt4(*(const v4f*)(qp + d0));
      h4 a1 = cvt4(*(const v4f*)(qp + d0 + 4));
      h4 a2 = cvt4(*(const v4f*)(qp + d0 + 8));
      h4 a3 = cvt4(*(const v4f*)(qp + d0 + 12));
      v8h lo = __builtin_shufflevector(a0, a1, 0, 1, 2, 3, 4, 5, 6, 7);
      v8h hi = __builtin_shufflevector(a2, a3, 0, 1, 2, 3, 4, 5, 6, 7);
      bq[c] = __builtin_shufflevector(lo, hi, 0, 1, 2, 3, 4, 5, 6, 7,
                                      8, 9, 10, 11, 12, 13, 14, 15);
    }
  }

  v8f o[4];
  #pragma unroll
  for (int t = 0; t < 4; ++t)
    #pragma unroll
    for (int r = 0; r < 8; ++r) o[t][r] = 0.0f;
  float m = -1e30f, lsum = 0.0f;
  const float scale = rsqrtf((float)DHc);

  const int nkb = (qblk * 64 + 64) / 32;
  for (int kb = 0; kb < nkb; ++kb) {
    const int ks = kb * 32;
    #pragma unroll
    for (int it = 0; it < 4; ++it) {
      int i = tid + it * 128;
      int r = i >> 4, c4 = i & 15;
      v4f kv = *(const v4f*)(kbase + (size_t)(ks + r) * row3D + c4 * 4);
      v4f vv = *(const v4f*)(vbase + (size_t)(ks + r) * row3D + c4 * 4);
      *(h4*)&Ks[r][c4 * 4] = cvt4(kv);
      *(h4*)&Vs[r][c4 * 4] = cvt4(vv);
    }
    __syncthreads();

    v8f s[2];
    #pragma unroll
    for (int t = 0; t < 2; ++t) {
      #pragma unroll
      for (int r = 0; r < 8; ++r) s[t][r] = 0.0f;
      #pragma unroll
      for (int c = 0; c < 2; ++c) {
        const _Float16* kr = &Ks[t * 16 + l16][c * 32 + half * 16];
        v16h ak = ld16(kr, kr + 8);
        s[t] = wmma_f16(ak, bq[c], s[t]);
      }
    }

    float mx = -1e30f;
    #pragma unroll
    for (int t = 0; t < 2; ++t)
      #pragma unroll
      for (int r = 0; r < 8; ++r) {
        float v = s[t][r] * scale;
        if (ks + t * 16 + half * 8 + r > qg) v = -1e30f;
        s[t][r] = v;
        mx = fmaxf(mx, v);
      }
    mx = fmaxf(mx, __shfl_xor(mx, 16, 32));
    float mnew = fmaxf(m, mx);
    float psum = 0.0f;
    #pragma unroll
    for (int t = 0; t < 2; ++t)
      #pragma unroll
      for (int r = 0; r < 8; ++r) {
        float p = __expf(s[t][r] - mnew);
        s[t][r] = p;
        psum += p;
      }
    psum += __shfl_xor(psum, 16, 32);
    float alpha = __expf(m - mnew);
    lsum = lsum * alpha + psum;
    m = mnew;
    #pragma unroll
    for (int t = 0; t < 4; ++t)
      #pragma unroll
      for (int r = 0; r < 8; ++r) o[t][r] *= alpha;

    #pragma unroll
    for (int t = 0; t < 2; ++t)
      *(v8h*)&PsT[wave][l16][t * 16 + half * 8] = cvt8(s[t]);
    asm volatile("s_wait_dscnt 0" ::: "memory");
    v16h pb = ld16(&PsT[wave][l16][half * 8],
                   &PsT[wave][l16][half * 8 + 16]);

    #pragma unroll
    for (int t = 0; t < 4; ++t) {
      v16h av;
      #pragma unroll
      for (int j = 0; j < 8; ++j) {
        av[j]     = Vs[half * 8 + j][t * 16 + l16];
        av[8 + j] = Vs[half * 8 + 16 + j][t * 16 + l16];
      }
      o[t] = wmma_f16(av, pb, o[t]);
    }
    __syncthreads();
  }

  float linv = 1.0f / lsum;
  float* op = out + (size_t)(b * S + qg) * Dm + h * DHc;
  #pragma unroll
  for (int t = 0; t < 4; ++t) {
    v4f f0, f1;
    #pragma unroll
    for (int r = 0; r < 4; ++r) { f0[r] = o[t][r] * linv; f1[r] = o[t][4 + r] * linv; }
    int dd = t * 16 + half * 8;
    *(v4f*)(op + dd)     = f0;
    *(v4f*)(op + dd + 4) = f1;
  }
}

// ---------------------------------------------------------------------------
__global__ __launch_bounds__(256) void k_add_ln(
    const float* __restrict__ x, const float* __restrict__ r,
    const float* __restrict__ g, const float* __restrict__ b,
    float* __restrict__ out, int Dm)
{
  const int row = blockIdx.x;
  const int tid = threadIdx.x;
  const size_t base = (size_t)row * Dm;
  float v0 = x[base + tid];
  float v1 = x[base + tid + 256];
  if (r) { v0 += r[base + tid]; v1 += r[base + tid + 256]; }
  float s = v0 + v1, ss = v0 * v0 + v1 * v1;
  #pragma unroll
  for (int d = 1; d < 32; d <<= 1) {
    s  += __shfl_xor(s,  d, 32);
    ss += __shfl_xor(ss, d, 32);
  }
  __shared__ float sb[8], qb[8];
  if ((tid & 31) == 0) { sb[tid >> 5] = s; qb[tid >> 5] = ss; }
  __syncthreads();
  float S = 0.0f, Q = 0.0f;
  #pragma unroll
  for (int w = 0; w < 8; ++w) { S += sb[w]; Q += qb[w]; }
  float mean = S / Dm;
  float var  = Q / Dm - mean * mean;
  float inv  = rsqrtf(var + 1e-5f);
  out[base + tid]       = (v0 - mean) * inv * g[tid] + b[tid];
  out[base + tid + 256] = (v1 - mean) * inv * g[tid + 256] + b[tid + 256];
}

// ---------------------------------------------------------------------------
__global__ __launch_bounds__(256) void k_router(
    const float* __restrict__ x, const float* __restrict__ Wr,
    const float* __restrict__ br, float* __restrict__ g,
    float* __restrict__ fcnt, float* __restrict__ psum, int Dm)
{
  const int tok  = blockIdx.x * 8 + (threadIdx.x >> 5);
  const int lane = threadIdx.x & 31;
  float acc[8] = {0, 0, 0, 0, 0, 0, 0, 0};
  const float* xp = x + (size_t)tok * Dm;
  for (int d = lane; d < Dm; d += 32) {
    float xv = xp[d];
    const float* wp = Wr + (size_t)d * 8;
    #pragma unroll
    for (int e = 0; e < 8; ++e) acc[e] += xv * wp[e];
  }
  #pragma unroll
  for (int d = 1; d < 32; d <<= 1)
    #pragma unroll
    for (int e = 0; e < 8; ++e) acc[e] += __shfl_xor(acc[e], d, 32);

  float mx = -1e30f;
  #pragma unroll
  for (int e = 0; e < 8; ++e) { acc[e] += br[e]; mx = fmaxf(mx, acc[e]); }
  float p[8], se = 0.0f;
  #pragma unroll
  for (int e = 0; e < 8; ++e) { p[e] = __expf(acc[e] - mx); se += p[e]; }
  #pragma unroll
  for (int e = 0; e < 8; ++e) p[e] /= se;

  int i0 = 0;
  #pragma unroll
  for (int e = 1; e < 8; ++e) if (p[e] > p[i0]) i0 = e;
  int i1 = (i0 == 0) ? 1 : 0;
  #pragma unroll
  for (int e = 0; e < 8; ++e) if (e != i0 && p[e] > p[i1]) i1 = e;
  float w0 = p[i0], w1 = p[i1], wsum = w0 + w1;
  w0 /= wsum; w1 /= wsum;

  if (lane == 0) {
    #pragma unroll
    for (int e = 0; e < 8; ++e)
      g[(size_t)tok * 8 + e] = (e == i0) ? w0 : ((e == i1) ? w1 : 0.0f);
    atomicAdd(&fcnt[i0], 1.0f);
    atomicAdd(&fcnt[i1], 1.0f);
    #pragma unroll
    for (int e = 0; e < 8; ++e) atomicAdd(&psum[e], p[e]);
  }
}

// ---------------------------------------------------------------------------
__global__ void k_embed(const int* __restrict__ tgt,
                        const float* __restrict__ emb,
                        float* __restrict__ x, int Dm)
{
  const int t = blockIdx.x, tid = threadIdx.x;
  const size_t src = (size_t)tgt[t] * Dm;
  x[(size_t)t * Dm + tid]       = emb[src + tid];
  x[(size_t)t * Dm + tid + 256] = emb[src + tid + 256];
}

__global__ void k_zero(float* __restrict__ p, int n) {
  int i = blockIdx.x * 256 + threadIdx.x;
  if (i < n) p[i] = 0.0f;
}

// stats: [0..7]=fcnt, [8..15]=psum, [16]=aux accum, [17..24]=usage accum
__global__ void k_aux(float* __restrict__ stats, float invMK, float invM) {
  float a = 0.0f;
  for (int e = 0; e < 8; ++e) {
    float f = stats[e] * invMK;
    float P = stats[8 + e] * invM;
    a += f * P;
    stats[17 + e] += f;
  }
  stats[16] += 8.0f * a;
}

__global__ void k_final(const float* __restrict__ stats,
                        float* __restrict__ out, int base, float invL) {
  out[base] = stats[16] * invL;
  for (int e = 0; e < 8; ++e) out[base + 1 + e] = stats[17 + e] * invL;
}

// ---------------------------------------------------------------------------
extern "C" void kernel_launch(void* const* d_in, const int* in_sizes, int n_in,
                              void* d_out, int out_size, void* d_ws, size_t ws_size,
                              hipStream_t stream)
{
  constexpr int Bc = 4, Sc = 2048, Dc = 512, Hc = 8, Fc = 2048;
  constexpr int Lc = 6, Ec = 8, Kc = 2, Vc = 118;
  constexpr int Mc = Bc * Sc;
  constexpr size_t MD = (size_t)Mc * Dc;
  constexpr size_t MF = (size_t)Mc * Fc;

  const int*   tgt   = (const int*)  d_in[0];
  const float* emb   = (const float*)d_in[1];
  const float* Wqkv  = (const float*)d_in[2];
  const float* bqkv  = (const float*)d_in[3];
  const float* Wo    = (const float*)d_in[4];
  const float* bo    = (const float*)d_in[5];
  const float* ln1g  = (const float*)d_in[6];
  const float* ln1b  = (const float*)d_in[7];
  const float* ln2g  = (const float*)d_in[8];
  const float* ln2b  = (const float*)d_in[9];
  const float* Wr    = (const float*)d_in[10];
  const float* br    = (const float*)d_in[11];
  const float* W1    = (const float*)d_in[12];
  const float* b1    = (const float*)d_in[13];
  const float* W2    = (const float*)d_in[14];
  const float* b2    = (const float*)d_in[15];
  const float* lnfg  = (const float*)d_in[16];
  const float* lnfb  = (const float*)d_in[17];
  const float* Wp    = (const float*)d_in[18];
  const float* bp    = (const float*)d_in[19];
  float* outp = (float*)d_out;

  float* ws    = (float*)d_ws;
  float* x     = ws;                     // MD
  float* qkvb  = ws + MD;                // 3*MD (reused for o-proj out)
  float* attn  = ws + 4 * MD;            // MD
  float* moe   = ws + 5 * MD;            // MD
  float* hbuf  = ws + 6 * MD;            // MF
  float* gbuf  = ws + 6 * MD + MF;       // M*E
  float* stats = gbuf + (size_t)Mc * Ec; // 25 floats

  dim3 thr128(128), thr256(256);

  k_zero<<<dim3(1), thr256, 0, stream>>>(stats, 25);
  k_embed<<<dim3(Mc), thr256, 0, stream>>>(tgt, emb, x, Dc);

  for (int l = 0; l < Lc; ++l) {
    const float* Wqkv_l = Wqkv + (size_t)l * Dc * 3 * Dc;
    const float* bqkv_l = bqkv + (size_t)l * 3 * Dc;
    const float* Wo_l   = Wo   + (size_t)l * Dc * Dc;
    const float* bo_l   = bo   + (size_t)l * Dc;

    k_gemm<false, 0, false, false>
        <<<dim3((3 * Dc) / BN, Mc / BM), thr256, 0, stream>>>(
            x, Wqkv_l, bqkv_l, qkvb, Mc, 3 * Dc, Dc, nullptr, 0);

    k_attn<<<dim3(Bc * Hc, Sc / 64), thr128, 0, stream>>>(qkvb, attn, Sc, Dc, Hc);

    k_gemm<false, 0, false, false>
        <<<dim3(Dc / BN, Mc / BM), thr256, 0, stream>>>(
            attn, Wo_l, bo_l, qkvb, Mc, Dc, Dc, nullptr, 0);

    k_add_ln<<<dim3(Mc), thr256, 0, stream>>>(
        x, qkvb, ln1g + (size_t)l * Dc, ln1b + (size_t)l * Dc, x, Dc);

    k_zero<<<dim3(1), thr256, 0, stream>>>(stats, 16);
    k_router<<<dim3(Mc / 8), thr256, 0, stream>>>(
        x, Wr + (size_t)l * Dc * Ec, br + (size_t)l * Ec,
        gbuf, stats, stats + 8, Dc);

    k_zero<<<dim3((int)((MD + 255) / 256)), thr256, 0, stream>>>(moe, (int)MD);
    for (int e = 0; e < Ec; ++e) {
      const float* W1e = W1 + ((size_t)l * Ec + e) * Dc * Fc;
      const float* b1e = b1 + ((size_t)l * Ec + e) * Fc;
      const float* W2e = W2 + ((size_t)l * Ec + e) * Fc * Dc;
      const float* b2e = b2 + ((size_t)l * Ec + e) * Dc;
      k_gemm<false, 1, false, false>
          <<<dim3(Fc / BN, Mc / BM), thr256, 0, stream>>>(
              x, W1e, b1e, hbuf, Mc, Fc, Dc, nullptr, 0);
      k_gemm<false, 0, true, true>
          <<<dim3(Dc / BN, Mc / BM), thr256, 0, stream>>>(
              hbuf, W2e, b2e, moe, Mc, Dc, Fc, gbuf + e, Ec);
    }

    k_add_ln<<<dim3(Mc), thr256, 0, stream>>>(
        x, moe, ln2g + (size_t)l * Dc, ln2b + (size_t)l * Dc, x, Dc);

    k_aux<<<dim3(1), dim3(1), 0, stream>>>(
        stats, 1.0f / (float)(Mc * Kc), 1.0f / (float)Mc);
  }

  k_add_ln<<<dim3(Mc), thr256, 0, stream>>>(x, nullptr, lnfg, lnfb, x, Dc);

  k_gemm<true, 0, false, false>
      <<<dim3((Vc + BN - 1) / BN, Mc / BM), thr256, 0, stream>>>(
          x, Wp, bp, outp, Mc, Vc, Dc, nullptr, 0);

  k_final<<<dim3(1), dim3(1), 0, stream>>>(stats, outp, Mc * Vc, 1.0f / (float)Lc);
}
